// e_gcl_sparse_30442728194254
// MI455X (gfx1250) — matrix operand order repaired and checked by reading, not compiled, not run on hardware
//
#include <hip/hip_runtime.h>
#include <hip/hip_bf16.h>

// ---- types for WMMA ----
typedef __attribute__((ext_vector_type(16))) _Float16 v16h;
typedef __attribute__((ext_vector_type(8)))  float    v8f;

union HFrag {
    v16h  v;
    uint4 q[2];
};

__device__ __forceinline__ float silu_f(float v) {
    return v * (1.0f / (1.0f + __expf(-v)));
}

__device__ __forceinline__ v8f wmma_f16(const HFrag& a, const HFrag& b, v8f c) {
    return __builtin_amdgcn_wmma_f32_16x16x32_f16(false, a.v, false, b.v,
                                                  (short)0, c, false, false);
}

#define BB 64
#define NN 64
#define HH 128
#define CUTOFF_R 5.0f

// ---------------------------------------------------------------------------
// Kernel 1: transpose + f32->f16 convert weights into [n][k] layout
// ranges: We2T[0,16384) Wc1T[16384,32768) Wn1T[32768,81920) Wn2T[81920,98304)
// ---------------------------------------------------------------------------
__global__ __launch_bounds__(256) void prep_weights(
    const float* __restrict__ We2, const float* __restrict__ Wc1,
    const float* __restrict__ Wn1, const float* __restrict__ Wn2,
    _Float16* __restrict__ We2T, _Float16* __restrict__ Wc1T,
    _Float16* __restrict__ Wn1T, _Float16* __restrict__ Wn2T)
{
    int tid = blockIdx.x * blockDim.x + threadIdx.x;
    if (tid < 16384) {
        int n = tid >> 7, k = tid & 127;
        We2T[n * 128 + k] = (_Float16)We2[k * 128 + n];
    } else if (tid < 32768) {
        int u = tid - 16384;
        int n = u >> 7, k = u & 127;
        Wc1T[n * 128 + k] = (_Float16)Wc1[k * 128 + n];
    } else if (tid < 81920) {
        int u = tid - 32768;
        int n = u / 384, k = u - n * 384;
        Wn1T[n * 384 + k] = (_Float16)Wn1[k * 128 + n];
    } else if (tid < 98304) {
        int u = tid - 81920;
        int n = u >> 7, k = u & 127;
        Wn2T[n * 128 + k] = (_Float16)Wn2[k * 128 + n];
    }
}

// ---------------------------------------------------------------------------
// Kernel 2: A1[bn,h] = h[bn]@We1[0:128, h] + be1[h] ; A2[bn,h] = h[bn]@We1[128:256, h]
// ---------------------------------------------------------------------------
__global__ __launch_bounds__(256) void precompute_A(
    const float* __restrict__ h, const float* __restrict__ We1,
    const float* __restrict__ be1,
    float* __restrict__ A1, float* __restrict__ A2)
{
    int tid = blockIdx.x * blockDim.x + threadIdx.x;
    if (tid >= BB * NN * HH) return;
    int bn = tid >> 7, hh = tid & 127;
    const float* hr = h + bn * HH;
    float a1 = be1[hh];
    float a2 = 0.0f;
#pragma unroll 4
    for (int k = 0; k < 128; ++k) {
        float hv = hr[k];
        a1 += hv * We1[k * 128 + hh];
        a2 += hv * We1[(128 + k) * 128 + hh];
    }
    A1[tid] = a1;
    A2[tid] = a2;
}

// ---------------------------------------------------------------------------
// Kernel 3: per-(b,j) edge pipeline.  256 threads = 8 waves.
//  e1 = silu(A1_j + A2_k + d2*wd)           [64 x 128]  -> LDS f16
//  e2 = silu(e1 @ We2 + be2)                (WMMA)      -> LDS f16, mi += cut*e2
//  g  = e2 @ Wc1  (WMMA);  phi_k = sum_h silu(cut*g + bc1)*Wc2
//  x_up = C * sum_k diff*phi*mask
// ---------------------------------------------------------------------------
__global__ __launch_bounds__(256) void edge_kernel(
    const float* __restrict__ x, const unsigned char* __restrict__ nm,
    const float* __restrict__ A1, const float* __restrict__ A2,
    const float* __restrict__ We1,
    const _Float16* __restrict__ We2T, const _Float16* __restrict__ Wc1T,
    const float* __restrict__ be2, const float* __restrict__ bc1,
    const float* __restrict__ Wc2,
    float* __restrict__ MI, float* __restrict__ out_x)
{
    __shared__ _Float16 sE[64][136];   // e1 tile (k, h), 16B-aligned rows
    __shared__ _Float16 sE2[64][136];  // e2 tile
    __shared__ float s_diff[64][3];
    __shared__ float s_d2[64];
    __shared__ float s_cut[64];        // cubic_cutoff(d) * mask
    __shared__ float s_mask[64];
    __shared__ float s_A1j[128];
    __shared__ float s_mi[128];
    __shared__ float s_phi[64];

    const int t  = threadIdx.x;
    const int b  = blockIdx.x >> 6;
    const int j  = blockIdx.x & 63;
    const int bj = b * NN + j;

    // ---- stage A: geometry, cutoff, A1_j, zero accumulators ----
    if (t < 64) {
        int k = t;
        float dx = x[bj * 3 + 0] - x[(b * NN + k) * 3 + 0];
        float dy = x[bj * 3 + 1] - x[(b * NN + k) * 3 + 1];
        float dz = x[bj * 3 + 2] - x[(b * NN + k) * 3 + 2];
        float d2 = dx * dx + dy * dy + dz * dz;
        float d  = sqrtf(d2);
        float mk = ((nm[bj] != 0) && (nm[b * NN + k] != 0) &&
                    (d < CUTOFF_R) && (k != j)) ? 1.0f : 0.0f;
        float cut;
        if (d <= 0.0f)          cut = 1.0f;
        else if (d >= CUTOFF_R) cut = 0.0f;
        else cut = 1.0f - 1.5f * d2 * (1.0f / (CUTOFF_R * CUTOFF_R))
                        + 0.5f * d2 * d * (1.0f / (CUTOFF_R * CUTOFF_R * CUTOFF_R));
        s_diff[k][0] = dx; s_diff[k][1] = dy; s_diff[k][2] = dz;
        s_d2[k]   = d2;
        s_cut[k]  = cut * mk;
        s_mask[k] = mk;
        s_phi[k]  = 0.0f;
    }
    if (t < 128) {
        s_A1j[t] = A1[bj * HH + t];
        s_mi[t]  = 0.0f;
    }
    __syncthreads();

    // ---- stage B: build e1 tile (silu of factorized layer-1) ----
    for (int idx = t; idx < 64 * 128; idx += 256) {
        int k  = idx >> 7;
        int hh = idx & 127;
        float v = s_A1j[hh] + A2[(b * NN + k) * HH + hh]
                + s_d2[k] * We1[256 * 128 + hh];
        sE[k][hh] = (_Float16)silu_f(v);
    }
    __syncthreads();

    const int w    = t >> 5;
    const int lane = t & 31;
    const int r    = w & 3;        // row tile (k edges 16r..16r+15)
    const int cg   = w >> 2;       // column group: cols [64*cg, 64*cg+64)
    const int hi   = lane >> 4;    // half-wave
    const int l15  = lane & 15;
    const int mrow = r * 16 + l15; // A-fragment source row

    // ---- stage C: GEMM1  e2 = silu(e1 @ We2 + be2) ----
    {
        v8f acc[4];
#pragma unroll
        for (int ci = 0; ci < 4; ++ci) acc[ci] = (v8f){0,0,0,0,0,0,0,0};
#pragma unroll
        for (int s = 0; s < 4; ++s) {
            int ka = s * 32 + hi * 16;
            HFrag a;
            const _Float16* pa = &sE[mrow][ka];
            a.q[0] = *reinterpret_cast<const uint4*>(pa);
            a.q[1] = *reinterpret_cast<const uint4*>(pa + 8);
            int kb = s * 32 + hi * 16;
#pragma unroll
            for (int ci = 0; ci < 4; ++ci) {
                int n = cg * 64 + ci * 16 + l15;
                const uint4* pb = reinterpret_cast<const uint4*>(We2T + n * 128 + kb);
                HFrag bf; bf.q[0] = pb[0]; bf.q[1] = pb[1];
                acc[ci] = wmma_f16(a, bf, acc[ci]);
            }
        }
#pragma unroll
        for (int ci = 0; ci < 4; ++ci) {
            int n = cg * 64 + ci * 16 + l15;
            float bias = be2[n];
            float pm = 0.0f;
#pragma unroll
            for (int i = 0; i < 8; ++i) {
                int m = r * 16 + i + hi * 8;          // C-layout row
                float v = silu_f(acc[ci][i] + bias);
                pm += s_cut[m] * v;                   // mi accumulation
                sE2[m][n] = (_Float16)v;
            }
            atomicAdd(&s_mi[n], pm);
        }
    }
    __syncthreads();

    // ---- stage D: GEMM2  g = e2 @ Wc1 ;  phi via silu(cut*g + bc1) @ Wc2 ----
    {
        v8f acc[4];
#pragma unroll
        for (int ci = 0; ci < 4; ++ci) acc[ci] = (v8f){0,0,0,0,0,0,0,0};
#pragma unroll
        for (int s = 0; s < 4; ++s) {
            int ka = s * 32 + hi * 16;
            HFrag a;
            const _Float16* pa = &sE2[mrow][ka];
            a.q[0] = *reinterpret_cast<const uint4*>(pa);
            a.q[1] = *reinterpret_cast<const uint4*>(pa + 8);
            int kb = s * 32 + hi * 16;
#pragma unroll
            for (int ci = 0; ci < 4; ++ci) {
                int n = cg * 64 + ci * 16 + l15;
                const uint4* pb = reinterpret_cast<const uint4*>(Wc1T + n * 128 + kb);
                HFrag bf; bf.q[0] = pb[0]; bf.q[1] = pb[1];
                acc[ci] = wmma_f16(a, bf, acc[ci]);
            }
        }
        float phip[8];
#pragma unroll
        for (int i = 0; i < 8; ++i) phip[i] = 0.0f;
#pragma unroll
        for (int ci = 0; ci < 4; ++ci) {
            int n = cg * 64 + ci * 16 + l15;
            float wc = Wc2[n];
            float bb = bc1[n];
#pragma unroll
            for (int i = 0; i < 8; ++i) {
                int m = r * 16 + i + hi * 8;
                float ph = silu_f(s_cut[m] * acc[ci][i] + bb);
                phip[i] += ph * wc;
            }
        }
#pragma unroll
        for (int i = 0; i < 8; ++i)
            atomicAdd(&s_phi[r * 16 + i + hi * 8], phip[i]);
    }
    __syncthreads();

    // ---- stage E: write mi, finish coordinates ----
    if (t < 128) MI[bj * HH + t] = s_mi[t];
    if (t < 3) {
        float sum = 0.0f;
        for (int k = 0; k < 64; ++k)
            sum += s_diff[k][t] * s_phi[k] * s_mask[k];
        float xo = x[bj * 3 + t] + sum * (1.0f / 63.0f);
        xo = fminf(1000.0f, fmaxf(-1000.0f, xo));
        out_x[bj * 3 + t] = xo;
    }
}

// ---------------------------------------------------------------------------
// Kernel 4: node MLP.  32 rows per block; h_new = h + (silu([h|mi|h0]@Wn1+bn1)@Wn2+bn2)
// ---------------------------------------------------------------------------
__global__ __launch_bounds__(256) void node_kernel(
    const float* __restrict__ h, const float* __restrict__ MI,
    const float* __restrict__ h0,
    const _Float16* __restrict__ Wn1T, const _Float16* __restrict__ Wn2T,
    const float* __restrict__ bn1, const float* __restrict__ bn2,
    float* __restrict__ out_h)
{
    __shared__ _Float16 sF[32][392];   // nfeat tile (row, 384), rows 16B-aligned
    __shared__ _Float16 sH[32][136];   // hidden tile

    const int t = threadIdx.x;
    const int R = blockIdx.x * 32;

    for (int idx = t; idx < 32 * 384; idx += 256) {
        int r = idx / 384;
        int c = idx - r * 384;
        int row = R + r;
        float v;
        if      (c < 128) v = h [row * 128 + c];
        else if (c < 256) v = MI[row * 128 + (c - 128)];
        else              v = h0[row * 128 + (c - 256)];
        sF[r][c] = (_Float16)v;
    }
    __syncthreads();

    const int w    = t >> 5;
    const int lane = t & 31;
    const int r    = w & 1;        // row tile (2 x 16 rows)
    const int cg   = w >> 1;       // col group 0..3 -> cols [32*cg, 32*cg+32)
    const int hi   = lane >> 4;
    const int l15  = lane & 15;
    const int mrow = r * 16 + l15;

    // GEMM1: hidden = silu(nfeat @ Wn1 + bn1), K = 384
    {
        v8f acc[2] = { (v8f){0,0,0,0,0,0,0,0}, (v8f){0,0,0,0,0,0,0,0} };
#pragma unroll
        for (int s = 0; s < 12; ++s) {
            int ka = s * 32 + hi * 16;
            HFrag a;
            const _Float16* pa = &sF[mrow][ka];
            a.q[0] = *reinterpret_cast<const uint4*>(pa);
            a.q[1] = *reinterpret_cast<const uint4*>(pa + 8);
            int kb = s * 32 + hi * 16;
#pragma unroll
            for (int ci = 0; ci < 2; ++ci) {
                int n = cg * 32 + ci * 16 + l15;
                const uint4* pb = reinterpret_cast<const uint4*>(Wn1T + n * 384 + kb);
                HFrag bf; bf.q[0] = pb[0]; bf.q[1] = pb[1];
                acc[ci] = wmma_f16(a, bf, acc[ci]);
            }
        }
#pragma unroll
        for (int ci = 0; ci < 2; ++ci) {
            int n = cg * 32 + ci * 16 + l15;
            float bias = bn1[n];
#pragma unroll
            for (int i = 0; i < 8; ++i) {
                int m = r * 16 + i + hi * 8;
                sH[m][n] = (_Float16)silu_f(acc[ci][i] + bias);
            }
        }
    }
    __syncthreads();

    // GEMM2: out = hidden @ Wn2 + bn2 + h (residual), K = 128
    {
        v8f acc[2] = { (v8f){0,0,0,0,0,0,0,0}, (v8f){0,0,0,0,0,0,0,0} };
#pragma unroll
        for (int s = 0; s < 4; ++s) {
            int ka = s * 32 + hi * 16;
            HFrag a;
            const _Float16* pa = &sH[mrow][ka];
            a.q[0] = *reinterpret_cast<const uint4*>(pa);
            a.q[1] = *reinterpret_cast<const uint4*>(pa + 8);
            int kb = s * 32 + hi * 16;
#pragma unroll
            for (int ci = 0; ci < 2; ++ci) {
                int n = cg * 32 + ci * 16 + l15;
                const uint4* pb = reinterpret_cast<const uint4*>(Wn2T + n * 128 + kb);
                HFrag bf; bf.q[0] = pb[0]; bf.q[1] = pb[1];
                acc[ci] = wmma_f16(a, bf, acc[ci]);
            }
        }
#pragma unroll
        for (int ci = 0; ci < 2; ++ci) {
            int n = cg * 32 + ci * 16 + l15;
            float bias = bn2[n];
#pragma unroll
            for (int i = 0; i < 8; ++i) {
                int m = r * 16 + i + hi * 8;
                int row = R + m;
                out_h[row * 128 + n] = acc[ci][i] + bias + h[row * 128 + n];
            }
        }
    }
}

// ---------------------------------------------------------------------------
extern "C" void kernel_launch(void* const* d_in, const int* in_sizes, int n_in,
                              void* d_out, int out_size, void* d_ws, size_t ws_size,
                              hipStream_t stream)
{
    const float*         h   = (const float*)d_in[0];
    const float*         x   = (const float*)d_in[1];
    const unsigned char* nm  = (const unsigned char*)d_in[2];
    const float*         h0  = (const float*)d_in[3];
    const float*         We1 = (const float*)d_in[4];
    const float*         be1 = (const float*)d_in[5];
    const float*         We2 = (const float*)d_in[6];
    const float*         be2 = (const float*)d_in[7];
    const float*         Wn1 = (const float*)d_in[8];
    const float*         bn1 = (const float*)d_in[9];
    const float*         Wn2 = (const float*)d_in[10];
    const float*         bn2 = (const float*)d_in[11];
    const float*         Wc1 = (const float*)d_in[12];
    const float*         bc1 = (const float*)d_in[13];
    const float*         Wc2 = (const float*)d_in[14];

    float* out_h = (float*)d_out;                 // [B,N,H]
    float* out_x = out_h + BB * NN * HH;          // [B,N,3]

    char* ws = (char*)d_ws;
    _Float16* We2T = (_Float16*)(ws);                       //  32 KB
    _Float16* Wc1T = (_Float16*)(ws + 32768);               //  32 KB
    _Float16* Wn1T = (_Float16*)(ws + 65536);               //  96 KB
    _Float16* Wn2T = (_Float16*)(ws + 163840);              //  32 KB
    float*    A1   = (float*)   (ws + 196608);              //   2 MB
    float*    A2   = (float*)   (ws + 196608 + 2097152);    //   2 MB
    float*    MI   = (float*)   (ws + 196608 + 4194304);    //   2 MB

    prep_weights<<<384, 256, 0, stream>>>(We2, Wc1, Wn1, Wn2,
                                          We2T, Wc1T, Wn1T, Wn2T);
    precompute_A<<<2048, 256, 0, stream>>>(h, We1, be1, A1, A2);
    edge_kernel<<<BB * NN, 256, 0, stream>>>(x, nm, A1, A2, We1,
                                             We2T, Wc1T, be2, bc1, Wc2,
                                             MI, out_x);
    node_kernel<<<(BB * NN) / 32, 256, 0, stream>>>(h, MI, h0,
                                                    Wn1T, Wn2T, bn1, bn2,
                                                    out_h);
}
